// MSPathSampler_24816321036790
// MI455X (gfx1250) — matrix operand order repaired and checked by reading, not compiled, not run on hardware
//
#include <hip/hip_runtime.h>
#include <hip/hip_bf16.h>
#include <math.h>

// ---------------- problem dims ----------------
constexpr int BSIZE = 1024;
constexpr int DIM   = 16384;
constexpr int HID   = 1024;
constexpr int MAXR  = 15;   // 2*R - 1
constexpr int NB    = 4;    // N-tiles per wave (register blocking)

// ---------------- workspace layout (bytes) ----------------
constexpr size_t WP_OFF   = 0;                                    // packed f16 W   (DIM x HID)   32 MB
constexpr size_t WTP_OFF  = WP_OFF  + (size_t)DIM * HID * 2;      // packed f16 W^T (HID x DIM)   32 MB
constexpr size_t XH_OFF   = WTP_OFF + (size_t)DIM * HID * 2;      // xh / yh f16 (BSIZE x DIM)    32 MB (reused)
constexpr size_t GRAD_OFF = XH_OFF  + (size_t)BSIZE * DIM * 2;    // grad f32 (BSIZE x DIM)       64 MB
constexpr size_t H_OFF    = GRAD_OFF + (size_t)BSIZE * DIM * 4;   // h f16 (BSIZE x HID)           2 MB
constexpr size_t SCAL_OFF = H_OFF   + (size_t)BSIZE * HID * 2;    // per-row scalars

typedef __attribute__((ext_vector_type(16))) _Float16 v16h;
typedef __attribute__((ext_vector_type(8)))  _Float16 v8h;
typedef __attribute__((ext_vector_type(8)))  float    v8f;

// ---------------- RNG helper (deterministic hash) ----------------
__device__ __forceinline__ unsigned hash3(unsigned a, unsigned b, unsigned c) {
    unsigned h = a * 0x9E3779B1u + b * 0x85EBCA77u + c * 0xC2B2AE3Du + 0x27D4EB2Fu;
    h ^= h >> 15; h *= 0x2C1B3C6Du;
    h ^= h >> 12; h *= 0x297A2D39u;
    h ^= h >> 15;
    return h;
}
__device__ __forceinline__ float u01(unsigned h) {
    return (float)(h >> 8) * (1.0f / 16777216.0f) + 1e-12f;
}

// ---------------- f32 -> f16 convert (x -> xh) ----------------
__global__ void __launch_bounds__(256) cvt_f16_kernel(const float* __restrict__ src,
                                                      _Float16* __restrict__ dst,
                                                      size_t n) {
    for (size_t i = blockIdx.x * 256ull + threadIdx.x; i < n;
         i += (size_t)gridDim.x * 256ull)
        dst[i] = (_Float16)src[i];
}

// ---------------- pack W (DIM x HID) into WMMA B-fragment tiles ----------------
// B tile (32x16) for k-tile kt, n-tile nt: lane L holds col n = nt*16 + (L&15),
// K offsets (L>>4)*16 + t for t=0..15, stored contiguously (512 halfs per tile).
__global__ void __launch_bounds__(256) pack_W_kernel(const float* __restrict__ W,
                                                     _Float16* __restrict__ Wp) {
    const size_t total = (size_t)DIM * HID;
    constexpr int tilesN = HID / 16;
    for (size_t idx = blockIdx.x * 256ull + threadIdx.x; idx < total;
         idx += (size_t)gridDim.x * 256ull) {
        int t    = (int)(idx & 15);
        int lane = (int)((idx >> 4) & 31);
        size_t tile = idx >> 9;
        int nt = (int)(tile % tilesN);
        int kt = (int)(tile / tilesN);
        int k = kt * 32 + ((lane >> 4) << 4) + t;     // over DIM
        int n = nt * 16 + (lane & 15);                // over HID
        Wp[idx] = (_Float16)W[(size_t)k * HID + n];
    }
}

// pack W^T (logical B' of shape HID x DIM): B'[k][n] = W[n][k]
__global__ void __launch_bounds__(256) pack_Wt_kernel(const float* __restrict__ W,
                                                      _Float16* __restrict__ Wtp) {
    const size_t total = (size_t)DIM * HID;
    constexpr int tilesN = DIM / 16;
    for (size_t idx = blockIdx.x * 256ull + threadIdx.x; idx < total;
         idx += (size_t)gridDim.x * 256ull) {
        int t    = (int)(idx & 15);
        int lane = (int)((idx >> 4) & 31);
        size_t tile = idx >> 9;
        int nt = (int)(tile % tilesN);
        int kt = (int)(tile / tilesN);
        int k = kt * 32 + ((lane >> 4) << 4) + t;     // over HID
        int n = nt * 16 + (lane & 15);                // over DIM
        Wtp[idx] = (_Float16)W[(size_t)n * HID + k];
    }
}

// ---------------- WMMA GEMM: C(MxN) = A(MxK,f16 row-major) * Bp(packed f16) ----
// Each wave: one 16-row strip x NB 16-col tiles; A fragment reused across NB wmma.
// mode 0: val = acc + bias[n]; atomicAdd(spsum[m], softplus(val)); if(hout) hout = (f16)sigmoid(val)
// mode 1: val = acc + bias[n]; outf[m*N+n] = val
__global__ void __launch_bounds__(256)
gemm_wmma_kernel(const _Float16* __restrict__ A, const _Float16* __restrict__ Bp,
                 int M, int N, int K, int mode,
                 const float* __restrict__ bias,
                 float* __restrict__ outf,
                 _Float16* __restrict__ hout,
                 float* __restrict__ spsum) {
    const int lane   = threadIdx.x & 31;
    const int wave   = threadIdx.x >> 5;     // wave32
    const int tilesN = N >> 4;
    const int strips = tilesN / NB;
    const int w   = blockIdx.x * 8 + wave;
    const int tm  = w / strips;
    const int tn0 = (w % strips) * NB;
    if (tm >= (M >> 4)) return;

    const int row  = (tm << 4) + (lane & 15);
    const int koff = (lane >> 4) << 4;       // 0 or 16
    const size_t arow = (size_t)row * K;
    const int ktiles = K >> 5;

    v8f acc[NB] = {};
    for (int kt = 0; kt < ktiles; ++kt) {
        const int k0 = kt << 5;
        // ---- A fragment: two contiguous 16B loads of f16 ----
        const _Float16* ap = A + arow + k0 + koff;
        v8h a0 = *(const v8h*)(ap);
        v8h a1 = *(const v8h*)(ap + 8);
        v16h a;
#pragma unroll
        for (int e = 0; e < 8; ++e) { a[e] = a0[e]; a[8 + e] = a1[e]; }
        // ---- NB B fragments, one wmma each (A reused) ----
        const _Float16* bp = Bp + ((size_t)kt * tilesN + tn0) * 512 + lane * 16;
        if (kt + 1 < ktiles) {
            __builtin_prefetch((const void*)(ap + 32), 0, 1);
            __builtin_prefetch((const void*)(bp + (size_t)tilesN * 512), 0, 1);
        }
#pragma unroll
        for (int j = 0; j < NB; ++j) {
            v16h b = *(const v16h*)(bp + (size_t)j * 512);
            acc[j] = __builtin_amdgcn_wmma_f32_16x16x32_f16(
                /*neg_a=*/false, a, /*neg_b=*/false, b,
                /*c_mod=*/(short)0, acc[j], /*reuse_a=*/false, /*reuse_b=*/false);
        }
    }

    // ---- epilogue: C layout: VGPR v -> M = v + 8*(lane>>4), N = lane&15 ----
    const int m0 = (tm << 4) + ((lane >> 4) << 3);
#pragma unroll
    for (int j = 0; j < NB; ++j) {
        const int n = ((tn0 + j) << 4) + (lane & 15);
        const float bn = bias[n];
#pragma unroll
        for (int v = 0; v < 8; ++v) {
            const int m = m0 + v;
            const float val = acc[j][v] + bn;
            if (mode == 0) {
                const float sp = (val > 20.0f) ? val : log1pf(__expf(val));
                atomicAdd(&spsum[m], sp);
                if (hout) hout[(size_t)m * N + n] =
                    (_Float16)(1.0f / (1.0f + __expf(-val)));
            } else {
                outf[(size_t)m * N + n] = val;
            }
        }
    }
}

// ---------------- per-row path sampler (Gumbel-max categorical) ----------------
__global__ void __launch_bounds__(256)
sampler_kernel(const float* __restrict__ x, const float* __restrict__ grad,
               const float* __restrict__ c, const int* __restrict__ radius,
               float* __restrict__ y, _Float16* __restrict__ yh,
               float* __restrict__ dotxc, float* __restrict__ dotyc,
               float* __restrict__ lZx, float* __restrict__ lZy,
               float* __restrict__ ltil) {
    const int b = blockIdx.x;
    const int t = threadIdx.x;
    __shared__ unsigned char bits[DIM];
    __shared__ float rf[256], rs[256], rd[256], re[256];
    __shared__ int ri[256];

    const float* xr = x + (size_t)b * DIM;
    const float* gr = grad + (size_t)b * DIM;

    // ---- pass 1: cache bits, dot(x,c), online logsumexp((0.5-x)*g) ----
    float dot = 0.0f, m = -1e30f, s = 0.0f;
    for (int i = t; i < DIM; i += 256) {
        const float xv = xr[i];
        bits[i] = (xv > 0.5f) ? 1 : 0;
        dot += xv * c[i];
        const float l = (0.5f - xv) * gr[i];
        if (l > m) { s = s * __expf(m - l) + 1.0f; m = l; }
        else       { s += __expf(l - m); }
    }
    rf[t] = m; rs[t] = s; rd[t] = dot;
    __syncthreads();
    for (int off = 128; off; off >>= 1) {
        if (t < off) {
            rd[t] += rd[t + off];
            const float mo = rf[t + off], so = rs[t + off];
            if (rf[t] >= mo) rs[t] += so * __expf(mo - rf[t]);
            else { rs[t] = rs[t] * __expf(rf[t] - mo) + so; rf[t] = mo; }
        }
        __syncthreads();
    }
    if (t == 0) { lZx[b] = rf[0] + __logf(rs[0]); dotxc[b] = rd[0]; }
    __syncthreads();

    // ---- sampling steps ----
    const int r = radius[b];
    for (int step = 0; step < MAXR; ++step) {
        if (step >= r) break;
        float bm = -1e30f; int bi = 0;
        for (int i = t; i < DIM; i += 256) {
            const float l = (bits[i] ? -0.5f : 0.5f) * gr[i];
            const float gum = -__logf(-__logf(u01(hash3((unsigned)b, (unsigned)step, (unsigned)i))));
            const float v = l + gum;
            if (v > bm) { bm = v; bi = i; }
        }
        rf[t] = bm; ri[t] = bi;
        __syncthreads();
        for (int off = 128; off; off >>= 1) {
            if (t < off && rf[t + off] > rf[t]) { rf[t] = rf[t + off]; ri[t] = ri[t + off]; }
            __syncthreads();
        }
        if (t == 0) bits[ri[0]] ^= 1;
        __syncthreads();
    }

    // ---- pass 2: write y (f32 + f16), dot(y,c), -sum g*(y-x), logsumexp((0.5-y)*g) ----
    float doty = 0.0f, lt = 0.0f;
    float m2 = -1e30f, s2 = 0.0f;
    for (int i = t; i < DIM; i += 256) {
        const float yv = (float)bits[i];
        y [(size_t)b * DIM + i] = yv;
        yh[(size_t)b * DIM + i] = (_Float16)yv;
        const float g = gr[i];
        doty += yv * c[i];
        lt += g * (yv - xr[i]);
        const float l = (0.5f - yv) * g;
        if (l > m2) { s2 = s2 * __expf(m2 - l) + 1.0f; m2 = l; }
        else        { s2 += __expf(l - m2); }
    }
    rf[t] = m2; rs[t] = s2; rd[t] = doty; re[t] = lt;
    __syncthreads();
    for (int off = 128; off; off >>= 1) {
        if (t < off) {
            rd[t] += rd[t + off];
            re[t] += re[t + off];
            const float mo = rf[t + off], so = rs[t + off];
            if (rf[t] >= mo) rs[t] += so * __expf(mo - rf[t]);
            else { rs[t] = rs[t] * __expf(rf[t] - mo) + so; rf[t] = mo; }
        }
        __syncthreads();
    }
    if (t == 0) {
        lZy[b]   = rf[0] + __logf(rs[0]);
        dotyc[b] = rd[0];
        ltil[b]  = -re[0];
    }
}

// ---------------- acceptance test ----------------
__global__ void __launch_bounds__(256)
accept_kernel(const float* __restrict__ spx, const float* __restrict__ spy,
              const float* __restrict__ dxc, const float* __restrict__ dyc,
              const float* __restrict__ lZx, const float* __restrict__ lZy,
              const float* __restrict__ lt, float* __restrict__ accv) {
    const int b = blockIdx.x * 256 + threadIdx.x;
    if (b < BSIZE) {
        const float sx = spx[b] + dxc[b];
        const float sy = spy[b] + dyc[b];
        const float la = fminf(sy - sx + lt[b] + lZx[b] - lZy[b], 0.0f);
        const float u = u01(hash3((unsigned)b, 0xACCEu, 0x5EEDu));
        accv[b] = (__expf(la) >= u) ? 1.0f : 0.0f;
    }
}

// ---------------- select: out = acc ? y(already in out) : x ----------------
__global__ void __launch_bounds__(256)
select_kernel(const float* __restrict__ x, const float* __restrict__ accv,
              float* __restrict__ out) {
    const size_t idx = blockIdx.x * 256ull + threadIdx.x;
    const int b = (int)(idx >> 14);   // DIM = 16384
    if (accv[b] == 0.0f) out[idx] = x[idx];
}

// ---------------- launch ----------------
extern "C" void kernel_launch(void* const* d_in, const int* in_sizes, int n_in,
                              void* d_out, int out_size, void* d_ws, size_t ws_size,
                              hipStream_t stream) {
    (void)in_sizes; (void)n_in; (void)out_size; (void)ws_size;
    const float* x      = (const float*)d_in[0];
    const float* W      = (const float*)d_in[1];
    const float* bh     = (const float*)d_in[2];
    const float* c      = (const float*)d_in[3];
    const int*   radius = (const int*)d_in[4];
    float* out = (float*)d_out;

    char* ws = (char*)d_ws;
    _Float16* Wp   = (_Float16*)(ws + WP_OFF);
    _Float16* Wtp  = (_Float16*)(ws + WTP_OFF);
    _Float16* xh   = (_Float16*)(ws + XH_OFF);   // reused as yh after GEMM1
    float*    grad = (float*)(ws + GRAD_OFF);
    _Float16* h    = (_Float16*)(ws + H_OFF);
    float*    spx  = (float*)(ws + SCAL_OFF);
    float*    spy  = spx  + BSIZE;
    float*    dxc  = spy  + BSIZE;
    float*    dyc  = dxc  + BSIZE;
    float*    lZx  = dyc  + BSIZE;
    float*    lZy  = lZx  + BSIZE;
    float*    lt   = lZy  + BSIZE;
    float*    accv = lt   + BSIZE;

    // zero atomic accumulators (deterministic across graph replays)
    (void)hipMemsetAsync(spx, 0, 2 * BSIZE * sizeof(float), stream);

    // one-time packs of W into WMMA B-fragment layout (L2-resident afterwards)
    pack_W_kernel <<<4096, 256, 0, stream>>>(W, Wp);
    pack_Wt_kernel<<<4096, 256, 0, stream>>>(W, Wtp);

    // x -> f16 (binary: exact)
    cvt_f16_kernel<<<4096, 256, 0, stream>>>(x, xh, (size_t)BSIZE * DIM);

    // GEMM1: pre = x@W + bh ; fused softplus-sum -> spx, sigmoid -> h (f16)
    gemm_wmma_kernel<<<(BSIZE / 16) * (HID / 16) / (8 * NB), 256, 0, stream>>>(
        xh, Wp, BSIZE, HID, DIM, 0, bh, nullptr, h, spx);

    // GEMM2: grad = h@W^T + c
    gemm_wmma_kernel<<<(BSIZE / 16) * (DIM / 16) / (8 * NB), 256, 0, stream>>>(
        h, Wtp, BSIZE, DIM, HID, 1, c, grad, nullptr, nullptr);

    // per-row sampler: writes y into d_out (+ f16 copy into xh buffer as yh)
    sampler_kernel<<<BSIZE, 256, 0, stream>>>(x, grad, c, radius,
                                              out, xh, dxc, dyc, lZx, lZy, lt);

    // GEMM3: pre_y = y@W + bh ; fused softplus-sum -> spy
    gemm_wmma_kernel<<<(BSIZE / 16) * (HID / 16) / (8 * NB), 256, 0, stream>>>(
        xh, Wp, BSIZE, HID, DIM, 0, bh, nullptr, nullptr, spy);

    // acceptance + final select (out already holds y)
    accept_kernel<<<(BSIZE + 255) / 256, 256, 0, stream>>>(spx, spy, dxc, dyc,
                                                           lZx, lZy, lt, accv);
    select_kernel<<<(int)(((size_t)BSIZE * DIM) / 256), 256, 0, stream>>>(x, accv, out);
}
